// MultiHeadAttention_62405874811505
// MI455X (gfx1250) — matrix operand order repaired and checked by reading, not compiled, not run on hardware
//
#include <hip/hip_runtime.h>
#include <hip/hip_bf16.h>

// ---------------------------------------------------------------------------
// Causal multi-head attention for MI455X (gfx1250), bf16 WMMA pipeline.
//   B=2, S=2048, D=1024, H=16, hd=64
// Stages:
//   0) convert X -> bf16; transpose W{q,k,v,o} -> bf16 [N][K]
//   1) fused QKV GEMM (WMMA bf16, f32 acc) -> Q[b,h,s,64], K[b,h,s,64], V^T[b,h,64,s]
//   2) flash attention per (b, h, 128-row q tile), TDM double-buffered K/V tiles
//   3) output projection GEMM -> fp32 out + bias
// ---------------------------------------------------------------------------

#define S_LEN 2048
#define D_MOD 1024
#define N_HEAD 16
#define HD 64
#define M_ROWS 4096              // B*S

typedef __attribute__((ext_vector_type(16))) __bf16 v16bf;
typedef __attribute__((ext_vector_type(8)))  float  v8f;

union Frag { v16bf v; uint4 q[2]; };

__device__ inline v8f wmma_bf16(const Frag& a, const Frag& b, v8f c) {
  // D(16x16,f32) = A(16x32,bf16) * B(32x16,bf16) + C
  return __builtin_amdgcn_wmma_f32_16x16x32_bf16(
      /*neg_a=*/false, a.v, /*neg_b=*/false, b.v,
      /*c_mod=*/(short)0, c, /*reuse_a=*/false, /*reuse_b=*/false);
}

// ---------------------------------------------------------------------------
// Tensor Data Mover: 2D tile (dwords) global -> LDS with row padding.
// LDS rows are padded 32dw data + 4dw pad (pad_interval code 4 = 32 dwords,
// pad_amount code 3 = 4 dwords) -> 72 bf16 per row, matching the fragment
// loaders' bank-conflict-free layout.
#if defined(__gfx1250__) && __has_builtin(__builtin_amdgcn_tensor_load_to_lds)
#define USE_TDM 1
#else
#define USE_TDM 0
#endif

#if USE_TDM
typedef __attribute__((ext_vector_type(4))) unsigned int u32x4;
typedef __attribute__((ext_vector_type(8))) int i32x8;
typedef __attribute__((ext_vector_type(4))) int i32x4;

__device__ inline void tdm_load_2d(unsigned lds_off, const __bf16* gptr,
                                   unsigned tensor_d0_dw, unsigned tensor_d1,
                                   unsigned long long stride0_dw) {
  union { u32x4 v; unsigned u[4]; } g0;
  unsigned long long ga = (unsigned long long)(const void*)gptr;
  g0.u[0] = 1u;                                             // count=1, load, user
  g0.u[1] = lds_off;                                        // lds_addr (bytes)
  g0.u[2] = (unsigned)ga;                                   // global_addr[31:0]
  g0.u[3] = (unsigned)((ga >> 32) & 0x01ffffffULL)          // global_addr[56:32]
            | (2u << 30);                                   // type = 2 (image)
  union { i32x8 v; unsigned long long q[4]; } g1;
  const unsigned long long dw0 =
      (2ull << 16)      // data_size = 4 bytes
    | (1ull << 20)      // pad_enable
    | (4ull << 22)      // pad_interval: 32 dwords
    | (3ull << 25);     // pad_amount: 4 dwords
  const unsigned long long td0 = tensor_d0_dw;
  g1.q[0] = dw0 | ((td0 & 0xffffull) << 48);                // [63:48]=tensor_dim0 lo
  g1.q[1] = (td0 >> 16)                                     // [79:64]=tensor_dim0 hi
          | ((unsigned long long)tensor_d1 << 16)           // [111:80]=tensor_dim1
          | (32ull << 48);                                  // [127:112]=tile_dim0=32dw
  g1.q[2] = 64ull                                           // [143:128]=tile_dim1=64
          | ((stride0_dw & 0xffffffffull) << 32);           // [191:160]=stride0 lo
  g1.q[3] = (stride0_dw >> 32) & 0xffffull;                 // [207:192]=stride0 hi
  i32x4 z4 = {0, 0, 0, 0};                                  // groups 2/3: <=2D tensor
  i32x8 z8 = {0, 0, 0, 0, 0, 0, 0, 0};                      // unused trailing group
  // 6-arg toolchain variant (clang-23 / therock-10.0 headers)
  __builtin_amdgcn_tensor_load_to_lds(g0.v, g1.v, z4, z4, z8, 0);
}
#endif

// ---------------------------------------------------------------------------
// Stage 0a: f32 -> bf16 elementwise
__global__ __launch_bounds__(256) void cvt_f32_bf16(const float* __restrict__ in,
                                                    __bf16* __restrict__ out, int n) {
  int i = blockIdx.x * blockDim.x + threadIdx.x;
  int stride = gridDim.x * blockDim.x;
  for (; i < n; i += stride) out[i] = (__bf16)in[i];
}

// Stage 0b: W[K][N] f32 -> Wt[N][K] bf16 (tiled transpose)
__global__ __launch_bounds__(256) void transpose_w_bf16(const float* __restrict__ w,
                                                        __bf16* __restrict__ wt) {
  __shared__ float tile[32][33];
  int x  = blockIdx.x * 32 + threadIdx.x;   // col n of W
  int y0 = blockIdx.y * 32;                 // row k of W
#pragma unroll
  for (int i = 0; i < 32; i += 8)
    tile[threadIdx.y + i][threadIdx.x] = w[(size_t)(y0 + threadIdx.y + i) * D_MOD + x];
  __syncthreads();
  int tx  = blockIdx.y * 32 + threadIdx.x;  // col k of Wt
  int ty0 = blockIdx.x * 32;                // row n of Wt
#pragma unroll
  for (int i = 0; i < 32; i += 8)
    wt[(size_t)(ty0 + threadIdx.y + i) * D_MOD + tx] = (__bf16)tile[threadIdx.x][threadIdx.y + i];
}

// ---------------------------------------------------------------------------
// Stage 1: fused QKV projection.
// C[m][n] = sum_k X[m][k] * Wt[n][k] + bias[n]; blockIdx.z selects Q/K/V.
// Block tile 128(M) x 64(N), 8 waves, each wave owns a 16x64 strip.
__global__ __launch_bounds__(256) void gemm_qkv(
    const __bf16* __restrict__ xb,
    const __bf16* __restrict__ wqt, const __bf16* __restrict__ wkt, const __bf16* __restrict__ wvt,
    const float* __restrict__ bq, const float* __restrict__ bk, const float* __restrict__ bv,
    __bf16* __restrict__ qb, __bf16* __restrict__ kb, __bf16* __restrict__ vt)
{
  const int z = blockIdx.z;
  const __bf16* wt   = (z == 0) ? wqt : (z == 1) ? wkt : wvt;
  const float*  bias = (z == 0) ? bq  : (z == 1) ? bk  : bv;
  const int n0 = blockIdx.x * 64;
  const int m0 = blockIdx.y * 128;

  __shared__ alignas(16) __bf16 sA[128 * 40];   // 128 rows x 32 k, pad to 40
  __shared__ alignas(16) __bf16 sB[64 * 40];    // 64 n-rows x 32 k, pad to 40

  const int tid = threadIdx.x, lane = tid & 31, wv = tid >> 5;
  const int mloc = lane & 15, half = lane >> 4;

  v8f acc[4] = {};

  for (int k0 = 0; k0 < D_MOD; k0 += 32) {
    __syncthreads();
#pragma unroll
    for (int i = 0; i < 2; ++i) {                 // 512 16B chunks of A
      int c = tid + i * 256, r = c >> 2, ch = c & 3;
      *(uint4*)&sA[r * 40 + ch * 8] =
          *(const uint4*)(xb + (size_t)(m0 + r) * D_MOD + k0 + ch * 8);
    }
    { int r = tid >> 2, ch = tid & 3;             // 256 16B chunks of B
      *(uint4*)&sB[r * 40 + ch * 8] =
          *(const uint4*)(wt + (size_t)(n0 + r) * D_MOD + k0 + ch * 8); }
    __syncthreads();

    // load all fragments first, then issue the 4 WMMAs back-to-back
    Frag a, bf4[4];
    a.q[0] = *(const uint4*)&sA[(wv * 16 + mloc) * 40 + half * 8];
    a.q[1] = *(const uint4*)&sA[(wv * 16 + mloc) * 40 + 16 + half * 8];
#pragma unroll
    for (int t = 0; t < 4; ++t) {
      bf4[t].q[0] = *(const uint4*)&sB[(t * 16 + mloc) * 40 + half * 8];
      bf4[t].q[1] = *(const uint4*)&sB[(t * 16 + mloc) * 40 + 16 + half * 8];
    }
#pragma unroll
    for (int t = 0; t < 4; ++t) acc[t] = wmma_bf16(a, bf4[t], acc[t]);
  }

#pragma unroll
  for (int t = 0; t < 4; ++t)
#pragma unroll
    for (int j = 0; j < 8; ++j) {
      int gm = m0 + wv * 16 + j + 8 * half;       // row in [0, B*S)
      int gn = n0 + t * 16 + mloc;                // col in [0, D)
      float vacc = acc[t][j] + bias[gn];
      int b = gm >> 11, s = gm & (S_LEN - 1);
      int h = gn >> 6,  dh = gn & (HD - 1);
      if (z == 0)
        qb[(((size_t)(b * N_HEAD + h) * S_LEN + s) << 6) + dh] = (__bf16)vacc;
      else if (z == 1)
        kb[(((size_t)(b * N_HEAD + h) * S_LEN + s) << 6) + dh] = (__bf16)vacc;
      else
        vt[((size_t)(b * N_HEAD + h) * HD + dh) * S_LEN + s] = (__bf16)vacc;   // V transposed
    }
}

// ---------------------------------------------------------------------------
// Stage 2: flash attention. Block = 128 q rows x full hd=64, 8 waves, each wave
// owns 16 complete q rows (wave-local online softmax). K/V tiles are staged by
// the Tensor Data Mover into double-buffered LDS, overlapped with compute.
__global__ __launch_bounds__(256) void attn_flash(
    const __bf16* __restrict__ qb, const __bf16* __restrict__ kb,
    const __bf16* __restrict__ vt, __bf16* __restrict__ ab)
{
  const int q0 = blockIdx.x * 128;
  const int h  = blockIdx.y;
  const int b  = blockIdx.z;
  const __bf16* Q = qb + ((size_t)(b * N_HEAD + h) * S_LEN) * HD;
  const __bf16* K = kb + ((size_t)(b * N_HEAD + h) * S_LEN) * HD;
  const __bf16* V = vt + ((size_t)(b * N_HEAD + h) * HD) * S_LEN;   // [hd][s]

  __shared__ alignas(16) __bf16 sK[2][64 * 72];   // [key][hd], rows padded to 72
  __shared__ alignas(16) __bf16 sV[2][64 * 72];   // [hd][key], rows padded to 72
  __shared__ alignas(16) __bf16 sP[128 * 72];     // [qrow][key], wave-private rows

  const int tid = threadIdx.x, lane = tid & 31, wv = tid >> 5;
  const int nloc = lane & 15, half = lane >> 4;

  // Q strip (16 rows x 64 hd) -> two A fragments, kept in registers
  Frag qa[2];
  { const __bf16* qr = Q + (size_t)(q0 + wv * 16 + nloc) * HD;
    qa[0].q[0] = *(const uint4*)(qr + half * 8);
    qa[0].q[1] = *(const uint4*)(qr + 16 + half * 8);
    qa[1].q[0] = *(const uint4*)(qr + 32 + half * 8);
    qa[1].q[1] = *(const uint4*)(qr + 48 + half * 8); }

  v8f o[4] = {};
  float m_i[8], l_i[8];
#pragma unroll
  for (int j = 0; j < 8; ++j) { m_i[j] = -3.0e38f; l_i[j] = 0.f; }

  const float scale = 0.125f;                    // 1/sqrt(64)
  const int nsteps = (q0 + 128) / 64;            // causal: keys <= q0+127

#if USE_TDM
  const unsigned kofs[2] = { (unsigned)(size_t)(void*)&sK[0][0],
                             (unsigned)(size_t)(void*)&sK[1][0] };
  const unsigned vofs[2] = { (unsigned)(size_t)(void*)&sV[0][0],
                             (unsigned)(size_t)(void*)&sV[1][0] };
  if (wv == 0) {                                 // prologue: DMA step 0 -> buf 0
    tdm_load_2d(kofs[0], K, 32, S_LEN, 32);      // K rows contiguous (32dw stride)
    tdm_load_2d(vofs[0], V, D_MOD, HD, 1024);    // V^T rows strided by S (1024dw)
  }
#endif

  for (int i = 0; i < nsteps; ++i) {
    const int cur = i & 1;
    const int kk = i * 64;
#if USE_TDM
    __syncthreads();                             // buf[cur^1] free for DMA reuse
    if (wv == 0) {
      if (i + 1 < nsteps) {
        const int kk1 = kk + 64;
        tdm_load_2d(kofs[cur ^ 1], K + (size_t)kk1 * HD, 32, S_LEN, 32);
        tdm_load_2d(vofs[cur ^ 1], V + kk1, D_MOD, HD, 1024);
        __builtin_amdgcn_s_wait_tensorcnt(2);    // step i done (in-order), i+1 in flight
      } else {
        __builtin_amdgcn_s_wait_tensorcnt(0);
      }
    }
    __syncthreads();                             // buf[cur] published to all waves
#else
    __syncthreads();
#pragma unroll
    for (int ii = 0; ii < 2; ++ii) {             // fallback staging
      int c = tid + ii * 256, r = c >> 3, ch = c & 7;
      *(uint4*)&sK[cur][r * 72 + ch * 8] = *(const uint4*)(K + (size_t)(kk + r) * HD + ch * 8);
      *(uint4*)&sV[cur][r * 72 + ch * 8] = *(const uint4*)(V + (size_t)r * S_LEN + kk + ch * 8);
    }
    __syncthreads();
#endif

    // scores: 16 q rows x 64 keys, f32
    float sc[4][8];
#pragma unroll
    for (int t = 0; t < 4; ++t) {
      Frag kb0, kb1;
      int krow = (t * 16 + nloc) * 72;
      kb0.q[0] = *(const uint4*)&sK[cur][krow + half * 8];
      kb0.q[1] = *(const uint4*)&sK[cur][krow + 16 + half * 8];
      kb1.q[0] = *(const uint4*)&sK[cur][krow + 32 + half * 8];
      kb1.q[1] = *(const uint4*)&sK[cur][krow + 48 + half * 8];
      v8f s = {};
      s = wmma_bf16(qa[0], kb0, s);
      s = wmma_bf16(qa[1], kb1, s);
      int kg = kk + t * 16 + nloc;
#pragma unroll
      for (int j = 0; j < 8; ++j) {
        int qg = q0 + wv * 16 + j + 8 * half;
        sc[t][j] = (kg <= qg) ? s[j] * scale : -3.0e38f;
      }
    }

    // online softmax (rows are wave-local; 16-lane shfl_xor reductions)
#pragma unroll
    for (int j = 0; j < 8; ++j) {
      float r = fmaxf(fmaxf(sc[0][j], sc[1][j]), fmaxf(sc[2][j], sc[3][j]));
#pragma unroll
      for (int mk = 1; mk < 16; mk <<= 1) r = fmaxf(r, __shfl_xor(r, mk, 32));
      float mn = fmaxf(m_i[j], r);
      float corr = __expf(m_i[j] - mn);
      m_i[j] = mn;
      float rs = 0.f;
#pragma unroll
      for (int t = 0; t < 4; ++t) { float p = __expf(sc[t][j] - mn); sc[t][j] = p; rs += p; }
#pragma unroll
      for (int mk = 1; mk < 16; mk <<= 1) rs += __shfl_xor(rs, mk, 32);
      l_i[j] = l_i[j] * corr + rs;
#pragma unroll
      for (int t = 0; t < 4; ++t) o[t][j] *= corr;
      int prow = wv * 16 + j + 8 * half;
#pragma unroll
      for (int t = 0; t < 4; ++t) sP[prow * 72 + t * 16 + nloc] = (__bf16)sc[t][j];
    }

    // O += P(16x64 keys) @ V(64 keys x 64 hd); sP rows are wave-private
#pragma unroll
    for (int kh = 0; kh < 2; ++kh) {
      Frag pa;
      int pbase = (wv * 16 + nloc) * 72 + kh * 32;
      pa.q[0] = *(const uint4*)&sP[pbase + half * 8];
      pa.q[1] = *(const uint4*)&sP[pbase + 16 + half * 8];
#pragma unroll
      for (int t = 0; t < 4; ++t) {
        Frag vb;
        int vrow = (t * 16 + nloc) * 72 + kh * 32;
        vb.q[0] = *(const uint4*)&sV[cur][vrow + half * 8];
        vb.q[1] = *(const uint4*)&sV[cur][vrow + 16 + half * 8];
        o[t] = wmma_bf16(pa, vb, o[t]);
      }
    }
  }

  // epilogue: normalize and scatter to [B,S,D] bf16 (head-interleaved cols)
#pragma unroll
  for (int j = 0; j < 8; ++j) {
    float inv = (l_i[j] > 0.f) ? 1.f / l_i[j] : 0.f;
    int srow = q0 + wv * 16 + j + 8 * half;
#pragma unroll
    for (int t = 0; t < 4; ++t) {
      int col = h * HD + t * 16 + nloc;
      ab[(size_t)(b * S_LEN + srow) * D_MOD + col] = (__bf16)(o[t][j] * inv);
    }
  }
}

// ---------------------------------------------------------------------------
// Stage 3: output projection, fp32 result + bias.
__global__ __launch_bounds__(256) void gemm_out(
    const __bf16* __restrict__ ab, const __bf16* __restrict__ wot,
    const float* __restrict__ bo, float* __restrict__ out)
{
  const int n0 = blockIdx.x * 64;
  const int m0 = blockIdx.y * 128;

  __shared__ alignas(16) __bf16 sA[128 * 40];
  __shared__ alignas(16) __bf16 sB[64 * 40];

  const int tid = threadIdx.x, lane = tid & 31, wv = tid >> 5;
  const int mloc = lane & 15, half = lane >> 4;

  v8f acc[4] = {};

  for (int k0 = 0; k0 < D_MOD; k0 += 32) {
    __syncthreads();
#pragma unroll
    for (int i = 0; i < 2; ++i) {
      int c = tid + i * 256, r = c >> 2, ch = c & 3;
      *(uint4*)&sA[r * 40 + ch * 8] =
          *(const uint4*)(ab + (size_t)(m0 + r) * D_MOD + k0 + ch * 8);
    }
    { int r = tid >> 2, ch = tid & 3;
      *(uint4*)&sB[r * 40 + ch * 8] =
          *(const uint4*)(wot + (size_t)(n0 + r) * D_MOD + k0 + ch * 8); }
    __syncthreads();

    Frag a, bf4[4];
    a.q[0] = *(const uint4*)&sA[(wv * 16 + mloc) * 40 + half * 8];
    a.q[1] = *(const uint4*)&sA[(wv * 16 + mloc) * 40 + 16 + half * 8];
#pragma unroll
    for (int t = 0; t < 4; ++t) {
      bf4[t].q[0] = *(const uint4*)&sB[(t * 16 + mloc) * 40 + half * 8];
      bf4[t].q[1] = *(const uint4*)&sB[(t * 16 + mloc) * 40 + 16 + half * 8];
    }
#pragma unroll
    for (int t = 0; t < 4; ++t) acc[t] = wmma_bf16(a, bf4[t], acc[t]);
  }

#pragma unroll
  for (int t = 0; t < 4; ++t)
#pragma unroll
    for (int j = 0; j < 8; ++j) {
      int gm = m0 + wv * 16 + j + 8 * half;
      int gn = n0 + t * 16 + mloc;
      out[(size_t)gm * D_MOD + gn] = acc[t][j] + bo[gn];
    }
}

// ---------------------------------------------------------------------------
extern "C" void kernel_launch(void* const* d_in, const int* in_sizes, int n_in,
                              void* d_out, int out_size, void* d_ws, size_t ws_size,
                              hipStream_t stream) {
  const float* X  = (const float*)d_in[0];
  const float* Wq = (const float*)d_in[1];
  const float* bq = (const float*)d_in[2];
  const float* Wk = (const float*)d_in[3];
  const float* bk = (const float*)d_in[4];
  const float* Wv = (const float*)d_in[5];
  const float* bv = (const float*)d_in[6];
  const float* Wo = (const float*)d_in[7];
  const float* bo = (const float*)d_in[8];
  // d_in[9] = p_sampling (unused; dropout=0 eval path)

  char* ws = (char*)d_ws;
  // Workspace layout (48 MB total):
  __bf16* xb  = (__bf16*)(ws + 0);              // X bf16       [4096][1024]   8 MB
  __bf16* wqt = (__bf16*)(ws + (8u  << 20));    // Wq^T bf16    [1024][1024]   2 MB
  __bf16* wkt = (__bf16*)(ws + (10u << 20));
  __bf16* wvt = (__bf16*)(ws + (12u << 20));
  __bf16* wot = (__bf16*)(ws + (14u << 20));
  __bf16* qb  = (__bf16*)(ws + (16u << 20));    // Q  [b,h,s,64]               8 MB
  __bf16* kb  = (__bf16*)(ws + (24u << 20));    // K  [b,h,s,64]               8 MB
  __bf16* vt  = (__bf16*)(ws + (32u << 20));    // V^T[b,h,64,s]               8 MB
  __bf16* ab  = (__bf16*)(ws + (40u << 20));    // attn out [b,s,D]            8 MB

  // Stage 0
  cvt_f32_bf16<<<4096, 256, 0, stream>>>(X, xb, M_ROWS * D_MOD);
  dim3 tb(32, 8), tg(32, 32);
  transpose_w_bf16<<<tg, tb, 0, stream>>>(Wq, wqt);
  transpose_w_bf16<<<tg, tb, 0, stream>>>(Wk, wkt);
  transpose_w_bf16<<<tg, tb, 0, stream>>>(Wv, wvt);
  transpose_w_bf16<<<tg, tb, 0, stream>>>(Wo, wot);

  // Stage 1: QKV projection (z selects Q/K/V)
  gemm_qkv<<<dim3(D_MOD / 64, M_ROWS / 128, 3), 256, 0, stream>>>(
      xb, wqt, wkt, wvt, bq, bk, bv, qb, kb, vt);

  // Stage 2: flash attention (TDM double-buffered K/V staging)
  attn_flash<<<dim3(S_LEN / 128, N_HEAD, 2), 256, 0, stream>>>(qb, kb, vt, ab);

  // Stage 3: output projection
  gemm_out<<<dim3(D_MOD / 64, M_ROWS / 128), 256, 0, stream>>>(ab, wot, bo, (float*)d_out);
}
